// my_gcn_2439541424223
// MI455X (gfx1250) — matrix operand order repaired and checked by reading, not compiled, not run on hardware
//
#include <hip/hip_runtime.h>
#include <hip/hip_bf16.h>

// Shapes (fixed by the reference): B*S = 48, N = 4096 nodes, C = 64, H = 128.
#define NBS   48
#define NN    4096
#define CC    64
#define HH    128

#define KCH   128                 // K chunk staged per TDM transfer
#define LROW  136                 // padded LDS row pitch in halves (128 + 8)
#define TILEH (128 * LROW)        // halves per staged 128xKCH tile (34816 B)

typedef _Float16 v16h __attribute__((ext_vector_type(16)));
typedef _Float16 v8h  __attribute__((ext_vector_type(8)));
typedef _Float16 v4h  __attribute__((ext_vector_type(4)));
typedef float    v8f  __attribute__((ext_vector_type(8)));
typedef unsigned int u32x4 __attribute__((ext_vector_type(4)));
typedef int          i32x8 __attribute__((ext_vector_type(8)));
typedef int          i32x4 __attribute__((ext_vector_type(4)));

static __device__ __forceinline__ v16h cat_v8h(v8h lo, v8h hi) {
  v16h r;
#pragma unroll
  for (int i = 0; i < 8; ++i) { r[i] = lo[i]; r[i + 8] = hi[i]; }
  return r;
}

// ---------------------------------------------------------------------------
// TDM: stage a 128-row x 128-half tile (row stride 4096 halves in memory) into
// LDS, padding +16B after each 256B row (-> 272B LDS pitch, bank-friendly).
// D# per CDNA5 ISA: group0 = count/lds/global/type,
// group1 = data_size, pad cfg, tensor dims/strides, tile dims.
// ---------------------------------------------------------------------------
static __device__ __forceinline__ void tdm_stage_tile(unsigned int lds_off,
                                                      const _Float16* gsrc) {
  unsigned long long ga = (unsigned long long)(uintptr_t)gsrc;
  u32x4 g0;
  g0[0] = 1u;                                            // count=1 (valid)
  g0[1] = lds_off;                                       // lds_addr (bytes)
  g0[2] = (unsigned int)ga;                              // global_addr[31:0]
  g0[3] = (unsigned int)((ga >> 32) & 0x01FFFFFFu)       // global_addr[56:32]
          | (2u << 30);                                  // type=2 ("image")
  i32x8 g1;
  g1[0] = (1 << 16)            // data_size = 2 bytes
        | (1 << 20)            // pad_enable
        | (5 << 22)            // pad_interval: every 64 DWORDs (256 B)
        | (3 << 25);           // pad_amount: 4 DWORDs (16 B)
  g1[1] = (int)((NN & 0xFFFFu) << 16);                   // tensor_dim0 lo16
  g1[2] = (int)((NN >> 16) | ((NN & 0xFFFFu) << 16));    // dim0 hi16 | dim1 lo16
  g1[3] = (int)((NN >> 16) | (128u << 16));              // dim1 hi16 | tile_dim0=128
  g1[4] = 128;                                           // tile_dim1 = 128 rows
  g1[5] = NN;                                            // tensor_dim0_stride lo32
  g1[6] = 0;
  g1[7] = 0;
  i32x4 z4 = (i32x4){0, 0, 0, 0};                        // 2D: groups 2/3 unused
  i32x8 z8 = (i32x8){0, 0, 0, 0, 0, 0, 0, 0};
  __builtin_amdgcn_tensor_load_to_lds(g0, g1, z4, z4, z8, 0);
}

// ---------------------------------------------------------------------------
// 1) d[i] = rsqrt(sum_j A[i][j] + 1)
// ---------------------------------------------------------------------------
__global__ __launch_bounds__(256) void k_rowsum(const float* __restrict__ A,
                                                float* __restrict__ deg) {
  const int row = blockIdx.x;
  const float* ar = A + (size_t)row * NN;
  float s = 0.0f;
  for (int j = threadIdx.x * 4; j < NN; j += 256 * 4) {
    float4 a = *(const float4*)(ar + j);
    s += a.x + a.y + a.z + a.w;
  }
  __shared__ float red[256];
  red[threadIdx.x] = s;
  __syncthreads();
  for (int off = 128; off > 0; off >>= 1) {
    if (threadIdx.x < off) red[threadIdx.x] += red[threadIdx.x + off];
    __syncthreads();
  }
  if (threadIdx.x == 0) deg[row] = rsqrtf(red[0] + 1.0f);
}

// ---------------------------------------------------------------------------
// 2) Lt[i][j] = (i==j) - d[i]*A[i][j]*d[j], stored f16 row-major [4096][4096]
// ---------------------------------------------------------------------------
__global__ __launch_bounds__(256) void k_build_lt(const float* __restrict__ A,
                                                  const float* __restrict__ deg,
                                                  _Float16* __restrict__ Lt) {
  size_t idx = ((size_t)blockIdx.x * 256 + threadIdx.x) * 4;  // < 4096*4096
  const int i = (int)(idx >> 12);
  const int j = (int)(idx & (NN - 1));
  const float di = deg[i];
  float4 a  = *(const float4*)(A + idx);
  float4 dj = *(const float4*)(deg + j);
  v4h o;
  o[0] = (_Float16)(((j + 0) == i ? 1.0f : 0.0f) - di * a.x * dj.x);
  o[1] = (_Float16)(((j + 1) == i ? 1.0f : 0.0f) - di * a.y * dj.y);
  o[2] = (_Float16)(((j + 2) == i ? 1.0f : 0.0f) - di * a.z * dj.z);
  o[3] = (_Float16)(((j + 3) == i ? 1.0f : 0.0f) - di * a.w * dj.w);
  *(v4h*)(Lt + idx) = o;
}

// ---------------------------------------------------------------------------
// 3) f32 -> f16 elementwise (count must be a multiple of 4)
// ---------------------------------------------------------------------------
__global__ __launch_bounds__(256) void k_cvt_f16(const float* __restrict__ src,
                                                 _Float16* __restrict__ dst,
                                                 int count) {
  int i = (blockIdx.x * 256 + threadIdx.x) * 4;
  if (i >= count) return;
  float4 a = *(const float4*)(src + i);
  v4h o;
  o[0] = (_Float16)a.x; o[1] = (_Float16)a.y;
  o[2] = (_Float16)a.z; o[3] = (_Float16)a.w;
  *(v4h*)(dst + i) = o;
}

// ---------------------------------------------------------------------------
// 4) Embedding GEMM via WMMA: xT[bs][h][n] = sum_c Wh[h][c] * xh[bs*N+n][c]
//    M=128 (h), 128-node chunk per block, K=64. 8 waves in 4(i)x2(j).
// ---------------------------------------------------------------------------
__global__ __launch_bounds__(256) void k_emb_gemm(const _Float16* __restrict__ Wh,
                                                  const _Float16* __restrict__ xh,
                                                  _Float16* __restrict__ xT) {
  const int lane = threadIdx.x & 31;
  const int w    = threadIdx.x >> 5;
  const int wi   = w & 3;
  const int wj   = w >> 2;
  const int bs   = blockIdx.y;
  const int n0   = blockIdx.x * 128 + wj * 64;
  const int m0   = wi * 32;
  const int mr   = lane & 15;
  const int kh   = lane >> 4;

  const _Float16* xb = xh + (size_t)bs * NN * CC;

  v8f acc[2][4];
#pragma unroll
  for (int it = 0; it < 2; ++it)
#pragma unroll
    for (int jt = 0; jt < 4; ++jt)
      acc[it][jt] = (v8f){0.f,0.f,0.f,0.f,0.f,0.f,0.f,0.f};

#pragma unroll
  for (int kk = 0; kk < CC; kk += 32) {
    v16h afr[2];
#pragma unroll
    for (int it = 0; it < 2; ++it) {
      const _Float16* ap = Wh + (size_t)(m0 + it * 16 + mr) * CC + kk + (kh << 3);
      afr[it] = cat_v8h(*(const v8h*)ap, *(const v8h*)(ap + 16));
    }
#pragma unroll
    for (int jt = 0; jt < 4; ++jt) {
      const _Float16* bp = xb + (size_t)(n0 + jt * 16 + mr) * CC + kk + (kh << 3);
      v16h bfr = cat_v8h(*(const v8h*)bp, *(const v8h*)(bp + 16));
#pragma unroll
      for (int it = 0; it < 2; ++it)
        acc[it][jt] = __builtin_amdgcn_wmma_f32_16x16x32_f16(
            false, afr[it], false, bfr, (short)0, acc[it][jt], false, false);
    }
  }

  _Float16* xo = xT + (size_t)bs * HH * NN;
  const int rhi = kh * 8;
#pragma unroll
  for (int it = 0; it < 2; ++it)
#pragma unroll
    for (int jt = 0; jt < 4; ++jt) {
      const int col = n0 + jt * 16 + mr;
#pragma unroll
      for (int r = 0; r < 8; ++r) {
        const int row = m0 + it * 16 + rhi + r;
        xo[(size_t)row * NN + col] = (_Float16)acc[it][jt][r];
      }
    }
}

// ---------------------------------------------------------------------------
// 5) Main GEMM + sigmoid, TDM double-buffered through LDS:
//    out[bs][i][j] = sigmoid( sum_k Lt[i][k] * xT[bs][j][k] )
//    Per block: 128x128 tile of H, K=4096 in 32 chunks of 128.
//    Wave 0 drives the Tensor Data Mover; all 8 waves eat from LDS.
// ---------------------------------------------------------------------------
__global__ __launch_bounds__(256) void k_main_gemm(const _Float16* __restrict__ Lt,
                                                   const _Float16* __restrict__ xT,
                                                   float* __restrict__ out) {
  extern __shared__ _Float16 smem[];   // 2 buffers x (Lt tile + xT tile)

  const int lane = threadIdx.x & 31;
  const int w    = threadIdx.x >> 5;
  const int wi   = w & 3;
  const int wj   = w >> 2;
  const int bs   = blockIdx.y;
  const int ib   = blockIdx.x * 128;     // block row base (node i)
  const int j0   = wj * 64;              // column group within 128 (h)
  const int mr   = lane & 15;
  const int kh   = lane >> 4;

  const _Float16* ltg = Lt + (size_t)ib * NN;           // Lt tile row base
  const _Float16* xtg = xT + (size_t)bs * HH * NN;      // xT row base (128 rows)
  const unsigned int lds_base = (unsigned int)(uintptr_t)smem;

  v8f acc[2][4];
#pragma unroll
  for (int it = 0; it < 2; ++it)
#pragma unroll
    for (int jt = 0; jt < 4; ++jt)
      acc[it][jt] = (v8f){0.f,0.f,0.f,0.f,0.f,0.f,0.f,0.f};

  // Prologue: stage chunk 0 into buffer 0.
  if (w == 0) {
    tdm_stage_tile(lds_base,                    ltg);
    tdm_stage_tile(lds_base + TILEH * 2,        xtg);
  }

  const int NCHUNK = NN / KCH;                  // 32
  for (int kc = 0; kc < NCHUNK; ++kc) {
    const int p = kc & 1;
    if (w == 0) {
      if (kc + 1 < NCHUNK) {
        const unsigned int nb = lds_base + (unsigned int)((p ^ 1) * (2 * TILEH) * 2);
        tdm_stage_tile(nb,             ltg + (kc + 1) * KCH);
        tdm_stage_tile(nb + TILEH * 2, xtg + (kc + 1) * KCH);
        __builtin_amdgcn_s_wait_tensorcnt(2);   // chunk kc's pair retired
      } else {
        __builtin_amdgcn_s_wait_tensorcnt(0);
      }
    }
    __syncthreads();                            // chunk kc visible to all waves

    const _Float16* ltile = smem + p * (2 * TILEH);
    const _Float16* xtile = ltile + TILEH;
#pragma unroll
    for (int ks = 0; ks < KCH / 32; ++ks) {
      const int kb = ks * 32;
      v16h afr[2];
#pragma unroll
      for (int it = 0; it < 2; ++it) {
        const _Float16* ap = ltile + (wi * 32 + it * 16 + mr) * LROW + kb + (kh << 3);
        afr[it] = cat_v8h(*(const v8h*)ap, *(const v8h*)(ap + 16));
      }
#pragma unroll
      for (int jt = 0; jt < 4; ++jt) {
        const _Float16* bp = xtile + (j0 + jt * 16 + mr) * LROW + kb + (kh << 3);
        v16h bfr = cat_v8h(*(const v8h*)bp, *(const v8h*)(bp + 16));
#pragma unroll
        for (int it = 0; it < 2; ++it)
          acc[it][jt] = __builtin_amdgcn_wmma_f32_16x16x32_f16(
              false, afr[it], false, bfr, (short)0, acc[it][jt], false, false);
      }
    }
    __syncthreads();                            // buffer reusable by next issue
  }

  const int rhi = kh * 8;
#pragma unroll
  for (int it = 0; it < 2; ++it)
#pragma unroll
    for (int jt = 0; jt < 4; ++jt) {
      const int col = j0 + jt * 16 + mr;
#pragma unroll
      for (int r = 0; r < 8; ++r) {
        const int row = ib + wi * 32 + it * 16 + rhi + r;
        const float v = acc[it][jt][r];
        out[((size_t)bs * NN + row) * HH + col] = 1.0f / (1.0f + __expf(-v));
      }
    }
}

// ---------------------------------------------------------------------------
// Launch
// ---------------------------------------------------------------------------
extern "C" void kernel_launch(void* const* d_in, const int* in_sizes, int n_in,
                              void* d_out, int out_size, void* d_ws, size_t ws_size,
                              hipStream_t stream) {
  const float* x    = (const float*)d_in[0];  // [48,4096,64]
  const float* A    = (const float*)d_in[1];  // [4096,4096]
  const float* Wemb = (const float*)d_in[2];  // [128,64]

  char* ws = (char*)d_ws;
  float*    d_deg = (float*)(ws + 0);                      // 16 KB
  _Float16* Wh    = (_Float16*)(ws + 16384);               // 16 KB
  _Float16* xT    = (_Float16*)(ws + 32768);               // 48 MB

  // Scratch staged inside d_out (stream-ordered; fully overwritten afterwards):
  //   head: out region -> temporarily x-f16 (25 MB); tail: A-copy region -> Lt-f16.
  char*     outB = (char*)d_out;
  _Float16* xh   = (_Float16*)outB;
  _Float16* Lt   = (_Float16*)(outB + (size_t)25165824 * 4);
  float*    out  = (float*)d_out;
  float*    Acpy = out + (size_t)25165824;

  const int xCount = NBS * NN * CC;   // 12,582,912
  const int wCount = HH * CC;         // 8,192

  k_rowsum<<<NN, 256, 0, stream>>>(A, d_deg);
  k_build_lt<<<(NN * NN) / (256 * 4), 256, 0, stream>>>(A, d_deg, Lt);
  k_cvt_f16<<<(xCount / 4 + 255) / 256, 256, 0, stream>>>(x, xh, xCount);
  k_cvt_f16<<<(wCount / 4 + 255) / 256, 256, 0, stream>>>(Wemb, Wh, wCount);
  k_emb_gemm<<<dim3(NN / 128, NBS), 256, 0, stream>>>(Wh, xh, xT);

  const size_t ldsBytes = (size_t)2 * 2 * TILEH * sizeof(_Float16);  // 139264
  k_main_gemm<<<dim3(NN / 128, NBS), 256, ldsBytes, stream>>>(Lt, xT, out);

  (void)hipMemcpyAsync(Acpy, A, (size_t)NN * NN * sizeof(float),
                       hipMemcpyDeviceToDevice, stream);
}
